// Transformer_16080357556777
// MI455X (gfx1250) — matrix operand order repaired and checked by reading, not compiled, not run on hardware
//
#include <hip/hip_runtime.h>

typedef __bf16 bf16;
typedef __bf16 v16bf __attribute__((ext_vector_type(16)));
typedef __bf16 v8bf  __attribute__((ext_vector_type(8)));
typedef float  v8f   __attribute__((ext_vector_type(8)));

#define BQ 512   // batch
#define TT 80    // seq len
#define DD 540   // d_model
#define HH 6     // heads
#define KK 90    // head dim
#define LL 6     // layers
#define DP 544   // padded D   (17 * 32)
#define KP 96    // padded head dim (6*16, 3*32)
#define SP 96    // padded seq for reduction (3*32)
#define HK 540   // H*K (ffn hidden)
#define HKP 544  // padded H*K

// ---------------------------------------------------------------------------
// WMMA fragment helpers (contiguous k = 16*half + e order in A and B, wave32)
// ---------------------------------------------------------------------------
__device__ __forceinline__ v8f wmma_bf16(v16bf a, v16bf b, v8f c) {
  return __builtin_amdgcn_wmma_f32_16x16x32_bf16(
      /*neg_a=*/false, a, /*neg_b=*/false, b,
      /*c_mod=*/(short)0, c, /*reuse_a=*/false, /*reuse_b=*/false);
}

// A: 16x32 bf16 tile from row-major [M, lda]; lane(0..15)=row m, half selects K group.
// => one contiguous 32B run at k-offset half*16.
__device__ __forceinline__ v16bf load_a_frag(const bf16* __restrict__ A, int lda,
                                             int mt, int kt, int lane) {
  int m    = mt * 16 + (lane & 15);
  int half = lane >> 4;
  const bf16* p = A + (size_t)m * lda + kt * 32 + half * 16;
  union { v16bf v; v8bf h[2]; } u;
  u.h[0] = *(const v8bf*)(p);
  u.h[1] = *(const v8bf*)(p + 8);
  return u.v;
}

// B: 32x16 bf16 tile where memory holds B^T row-major [N, ldb] (col n contiguous in k).
// lane: n = lane&15 ; element e: k = half*16 + e  => one contiguous 32B run.
__device__ __forceinline__ v16bf load_bt_frag(const bf16* __restrict__ Bt, int ldb,
                                              int nt, int kt, int lane) {
  int n    = nt * 16 + (lane & 15);
  int half = lane >> 4;
  const bf16* p = Bt + (size_t)n * ldb + kt * 32 + half * 16;
  union { v16bf v; v8bf h[2]; } u;
  u.h[0] = *(const v8bf*)(p);
  u.h[1] = *(const v8bf*)(p + 8);
  return u.v;
}

// B from prepacked weights: [ntile][ktile][lane][16] -> fully contiguous per lane.
__device__ __forceinline__ v16bf load_wp_frag(const bf16* __restrict__ Wp, int ktiles,
                                              int nt, int kt, int lane) {
  const bf16* p = Wp + ((((size_t)nt * ktiles + kt) << 5) + lane) * 16;
  union { v16bf v; v8bf h[2]; } u;
  u.h[0] = *(const v8bf*)(p);
  u.h[1] = *(const v8bf*)(p + 8);
  return u.v;
}

// ---------------------------------------------------------------------------
// Weight packing: logical fp32 [G][Kdim][N] -> bf16 [G][Ntiles][Ktiles][32][16],
// zero padded, matching the B-fragment lane layout above.
// ---------------------------------------------------------------------------
__global__ void pack_w_kernel(const float* __restrict__ src, bf16* __restrict__ dst,
                              int G, int Kdim, int N, int Ktiles, int Ntiles) {
  size_t total = (size_t)G * Ntiles * Ktiles * 512;
  size_t p = blockIdx.x * (size_t)blockDim.x + threadIdx.x;
  if (p >= total) return;
  int e    = (int)(p & 15);
  int lane = (int)((p >> 4) & 31);
  size_t rest = p >> 9;
  int kt = (int)(rest % Ktiles); rest /= Ktiles;
  int nt = (int)(rest % Ntiles); rest /= Ntiles;
  int g  = (int)rest;
  int n = nt * 16 + (lane & 15);
  int k = kt * 32 + (lane >> 4) * 16 + e;
  float v = (k < Kdim && n < N) ? src[((size_t)g * Kdim + k) * N + n] : 0.0f;
  dst[p] = (bf16)v;
}

// ---------------------------------------------------------------------------
// Embedding: x = embed[tok] + pos ; write fp32 [B,T,540] + padded bf16 [B,T,544]
// ---------------------------------------------------------------------------
__global__ void embed_kernel(const int* __restrict__ tok, const float* __restrict__ Emb,
                             const float* __restrict__ Pos, float* __restrict__ Xf,
                             bf16* __restrict__ Xb) {
  size_t idx = blockIdx.x * (size_t)blockDim.x + threadIdx.x;
  if (idx >= (size_t)BQ * TT * DP) return;
  int d = (int)(idx % DP);
  size_t bt = idx / DP;
  int t = (int)(bt % TT);
  if (d < DD) {
    int tk = tok[bt];
    float v = Emb[(size_t)tk * DD + d] + Pos[(size_t)t * DD + d];
    Xf[bt * DD + d] = v;
    Xb[bt * DP + d] = (bf16)v;
  } else {
    Xb[bt * DP + d] = (bf16)0.0f;
  }
}

// ---------------------------------------------------------------------------
// Fused QKV: one wave computes the q, k AND v 16x16 tiles for (b,h,mt,nt).
// Each A fragment feeds 3 independent WMMA chains (A traffic /3, hazards hidden).
//   q,k -> row-major [bh][T][KP] bf16 ; v -> transposed [bh][KP][SP] bf16
// ---------------------------------------------------------------------------
__global__ void gemm_qkv_fused_kernel(const bf16* __restrict__ X,
                                      const bf16* __restrict__ Wq,
                                      const bf16* __restrict__ Wk,
                                      const bf16* __restrict__ Wv,
                                      bf16* __restrict__ Q, bf16* __restrict__ Ko,
                                      bf16* __restrict__ Vt) {
  int wid  = blockIdx.x * (blockDim.x >> 5) + (threadIdx.x >> 5);
  int lane = threadIdx.x & 31;
  const int tilesPerBH = 5 * 6;
  if (wid >= BQ * HH * tilesPerBH) return;
  int bh = wid / tilesPerBH, tl = wid % tilesPerBH;
  int mt = tl / 6, nt = tl % 6;
  int b = bh / HH, h = bh % HH;
  const bf16* A  = X + (size_t)b * TT * DP;
  const size_t hoff = (size_t)h * (6 * 17 * 512);
  const bf16* wq = Wq + hoff;
  const bf16* wk = Wk + hoff;
  const bf16* wv = Wv + hoff;
  v8f aq = {}, ak = {}, av = {};
  for (int kt = 0; kt < 17; ++kt) {
    if (kt + 1 < 17)
      __builtin_prefetch((const void*)(A + (size_t)(mt * 16 + (lane & 15)) * DP + (kt + 1) * 32), 0, 3);
    v16bf a = load_a_frag(A, DP, mt, kt, lane);
    aq = wmma_bf16(a, load_wp_frag(wq, 17, nt, kt, lane), aq);
    ak = wmma_bf16(a, load_wp_frag(wk, 17, nt, kt, lane), ak);
    av = wmma_bf16(a, load_wp_frag(wv, 17, nt, kt, lane), av);
  }
  int n = nt * 16 + (lane & 15);
  int half = lane >> 4;
  bf16* oq = Q  + (size_t)bh * TT * KP;
  bf16* ok = Ko + (size_t)bh * TT * KP;
  bf16* ov = Vt + (size_t)bh * KP * SP;
#pragma unroll
  for (int r = 0; r < 8; ++r) {
    int m = mt * 16 + half * 8 + r;
    oq[(size_t)m * KP + n] = (bf16)aq[r];
    ok[(size_t)m * KP + n] = (bf16)ak[r];
    ov[(size_t)n * SP + m] = (bf16)av[r];   // transposed for attn@v B operand
  }
}

// scores: one wave computes a full 16x80 row block (5 N-tiles), A reused 5x.
__global__ void gemm_scores_kernel(const bf16* __restrict__ Q, const bf16* __restrict__ Kt,
                                   float* __restrict__ S) {
  int wid  = blockIdx.x * (blockDim.x >> 5) + (threadIdx.x >> 5);
  int lane = threadIdx.x & 31;
  if (wid >= BQ * HH * 5) return;
  int bh = wid / 5, mt = wid % 5;
  const bf16* A  = Q  + (size_t)bh * TT * KP;
  const bf16* Bt = Kt + (size_t)bh * TT * KP;
  v8f acc[5];
#pragma unroll
  for (int nt = 0; nt < 5; ++nt) acc[nt] = (v8f){};
  for (int kt = 0; kt < 3; ++kt) {
    v16bf a = load_a_frag(A, KP, mt, kt, lane);
#pragma unroll
    for (int nt = 0; nt < 5; ++nt)
      acc[nt] = wmma_bf16(a, load_bt_frag(Bt, KP, nt, kt, lane), acc[nt]);
  }
  float* o = S + (size_t)bh * TT * TT;
  int half = lane >> 4;
#pragma unroll
  for (int nt = 0; nt < 5; ++nt) {
    int n = nt * 16 + (lane & 15);
#pragma unroll
    for (int r = 0; r < 8; ++r) {
      int m = mt * 16 + half * 8 + r;
      o[(size_t)m * TT + n] = acc[nt][r];
    }
  }
}

// softmax over the QUERY axis (axis=2): column softmax of each [T,T] score matrix.
__global__ void softmax_kernel(const float* __restrict__ S, bf16* __restrict__ Attn) {
  int idx = blockIdx.x * blockDim.x + threadIdx.x;
  if (idx >= BQ * HH * SP) return;
  int s = idx % SP, bh = idx / SP;
  bf16* a = Attn + (size_t)bh * TT * SP + s;
  if (s >= TT) {
    for (int t = 0; t < TT; ++t) a[(size_t)t * SP] = (bf16)0.0f;
    return;
  }
  const float scale = 0.105409255339f;  // 90^-0.5
  const float* sc = S + (size_t)bh * TT * TT + s;
  float mx = -1e30f;
  for (int t = 0; t < TT; ++t) mx = fmaxf(mx, sc[(size_t)t * TT] * scale);
  float sum = 0.0f;
  for (int t = 0; t < TT; ++t) sum += __expf(sc[(size_t)t * TT] * scale - mx);
  float inv = 1.0f / sum;
  for (int t = 0; t < TT; ++t)
    a[(size_t)t * SP] = (bf16)(__expf(sc[(size_t)t * TT] * scale - mx) * inv);
}

// o = attn @ v : one wave computes a full 16x96 row block (6 N-tiles), 18 WMMAs.
__global__ void gemm_av_kernel(const bf16* __restrict__ Attn, const bf16* __restrict__ Vt,
                               float* __restrict__ O) {
  int wid  = blockIdx.x * (blockDim.x >> 5) + (threadIdx.x >> 5);
  int lane = threadIdx.x & 31;
  if (wid >= BQ * HH * 5) return;
  int bh = wid / 5, mt = wid % 5;
  int b = bh / HH, h = bh % HH;
  const bf16* A  = Attn + (size_t)bh * TT * SP;
  const bf16* Bt = Vt   + (size_t)bh * KP * SP;
  v8f acc[6];
#pragma unroll
  for (int nt = 0; nt < 6; ++nt) acc[nt] = (v8f){};
  for (int kt = 0; kt < 3; ++kt) {
    v16bf a = load_a_frag(A, SP, mt, kt, lane);
#pragma unroll
    for (int nt = 0; nt < 6; ++nt)
      acc[nt] = wmma_bf16(a, load_bt_frag(Bt, SP, nt, kt, lane), acc[nt]);
  }
  int half = lane >> 4;
#pragma unroll
  for (int nt = 0; nt < 6; ++nt) {
    int n = nt * 16 + (lane & 15);
    if (n < KK) {
#pragma unroll
      for (int r = 0; r < 8; ++r) {
        int m = mt * 16 + half * 8 + r;
        O[((size_t)b * TT + m) * DD + h * KK + n] = acc[nt][r];
      }
    }
  }
}

// LayerNorm(res + x): one wave32 per row of 540; fixed-trip unrolled loops keep
// the row cache in VGPRs (no scratch spill).
__global__ void ln_kernel(const float* __restrict__ Res, float* __restrict__ Xf,
                          bf16* __restrict__ Xb, const float* __restrict__ g,
                          const float* __restrict__ bb) {
  int row  = blockIdx.x * (blockDim.x >> 5) + (threadIdx.x >> 5);
  int lane = threadIdx.x & 31;
  if (row >= BQ * TT) return;
  const float* r = Res + (size_t)row * DD;
  float* x = Xf + (size_t)row * DD;
  float vals[17];
  float sum = 0.0f;
#pragma unroll
  for (int j = 0; j < 17; ++j) {
    int i = lane + j * 32;
    float v = (i < DD) ? (r[i] + x[i]) : 0.0f;
    vals[j] = v;
    sum += v;
  }
  for (int off = 16; off > 0; off >>= 1) sum += __shfl_xor(sum, off, 32);
  float mean = sum * (1.0f / DD);
  float var = 0.0f;
#pragma unroll
  for (int j = 0; j < 17; ++j) {
    int i = lane + j * 32;
    float d = (i < DD) ? (vals[j] - mean) : 0.0f;
    var += d * d;
  }
  for (int off = 16; off > 0; off >>= 1) var += __shfl_xor(var, off, 32);
  float inv = rsqrtf(var * (1.0f / DD) + 1e-5f);
  bf16* xb = Xb + (size_t)row * DP;
#pragma unroll
  for (int j = 0; j < 17; ++j) {
    int i = lane + j * 32;
    if (i < DD) {
      float y = (vals[j] - mean) * inv * g[i] + bb[i];
      x[i] = y;
      xb[i] = (bf16)y;
    }
  }
  if (lane < DP - DD) xb[DD + lane] = (bf16)0.0f;
}

// FFN GEMM 1: relu(x @ W1 + b1) -> bf16 [B,T,544] ; 2 N-tiles per wave.
__global__ void gemm_ffn1_kernel(const bf16* __restrict__ X, const bf16* __restrict__ Wp,
                                 const float* __restrict__ bias, bf16* __restrict__ Hout) {
  int wid  = blockIdx.x * (blockDim.x >> 5) + (threadIdx.x >> 5);
  int lane = threadIdx.x & 31;
  const int tilesPerB = 5 * 17;  // mt x (34/2) n-pairs
  if (wid >= BQ * tilesPerB) return;
  int b = wid / tilesPerB, tl = wid % tilesPerB;
  int mt = tl / 17, np = tl % 17;
  int nt0 = 2 * np, nt1 = nt0 + 1;
  const bf16* A = X + (size_t)b * TT * DP;
  v8f a0 = {}, a1 = {};
  for (int kt = 0; kt < 17; ++kt) {
    if (kt + 1 < 17)
      __builtin_prefetch((const void*)(A + (size_t)(mt * 16 + (lane & 15)) * DP + (kt + 1) * 32), 0, 3);
    v16bf a = load_a_frag(A, DP, mt, kt, lane);
    a0 = wmma_bf16(a, load_wp_frag(Wp, 17, nt0, kt, lane), a0);
    a1 = wmma_bf16(a, load_wp_frag(Wp, 17, nt1, kt, lane), a1);
  }
  int half = lane >> 4;
  bf16* o = Hout + (size_t)b * TT * HKP;
  int n0 = nt0 * 16 + (lane & 15), n1 = nt1 * 16 + (lane & 15);
  float bv0 = (n0 < HK) ? bias[n0] : 0.0f;
  float bv1 = (n1 < HK) ? bias[n1] : 0.0f;
#pragma unroll
  for (int r = 0; r < 8; ++r) {
    int m = mt * 16 + half * 8 + r;
    o[(size_t)m * HKP + n0] = (bf16)fmaxf(a0[r] + bv0, 0.0f);
    o[(size_t)m * HKP + n1] = (bf16)fmaxf(a1[r] + bv1, 0.0f);
  }
}

// FFN GEMM 2: h @ W2 + b2 -> f32 [B,T,540] ; 2 N-tiles per wave.
__global__ void gemm_ffn2_kernel(const bf16* __restrict__ Hin, const bf16* __restrict__ Wp,
                                 const float* __restrict__ bias, float* __restrict__ Fout) {
  int wid  = blockIdx.x * (blockDim.x >> 5) + (threadIdx.x >> 5);
  int lane = threadIdx.x & 31;
  const int tilesPerB = 5 * 17;
  if (wid >= BQ * tilesPerB) return;
  int b = wid / tilesPerB, tl = wid % tilesPerB;
  int mt = tl / 17, np = tl % 17;
  int nt0 = 2 * np, nt1 = nt0 + 1;
  const bf16* A = Hin + (size_t)b * TT * HKP;
  v8f a0 = {}, a1 = {};
  for (int kt = 0; kt < 17; ++kt) {
    v16bf a = load_a_frag(A, HKP, mt, kt, lane);
    a0 = wmma_bf16(a, load_wp_frag(Wp, 17, nt0, kt, lane), a0);
    a1 = wmma_bf16(a, load_wp_frag(Wp, 17, nt1, kt, lane), a1);
  }
  int half = lane >> 4;
  int n0 = nt0 * 16 + (lane & 15), n1 = nt1 * 16 + (lane & 15);
#pragma unroll
  for (int r = 0; r < 8; ++r) {
    int m = mt * 16 + half * 8 + r;
    if (n0 < DD) Fout[((size_t)b * TT + m) * DD + n0] = a0[r] + bias[n0];
    if (n1 < DD) Fout[((size_t)b * TT + m) * DD + n1] = a1[r] + bias[n1];
  }
}

// classifier: out[b] = x_flat . Wc + bc ; one 256-thread block per b.
__global__ void final_kernel(const float* __restrict__ Xf, const float* __restrict__ Wc,
                             const float* __restrict__ bc, float* __restrict__ Out) {
  int b = blockIdx.x;
  const float* x = Xf + (size_t)b * (TT * DD);
  float sum = 0.0f;
  for (int i = threadIdx.x; i < TT * DD; i += blockDim.x) sum += x[i] * Wc[i];
  for (int off = 16; off > 0; off >>= 1) sum += __shfl_xor(sum, off, 32);
  __shared__ float red[8];
  int w = threadIdx.x >> 5;
  if ((threadIdx.x & 31) == 0) red[w] = sum;
  __syncthreads();
  if (threadIdx.x == 0) {
    float t = 0.0f;
    for (int i = 0; i < 8; ++i) t += red[i];
    Out[b] = t + bc[0];
  }
}

// ---------------------------------------------------------------------------
extern "C" void kernel_launch(void* const* d_in, const int* in_sizes, int n_in,
                              void* d_out, int out_size, void* d_ws, size_t ws_size,
                              hipStream_t stream) {
  (void)in_sizes; (void)n_in; (void)out_size; (void)ws_size;
  const int*   tok = (const int*)  d_in[0];
  const float* emb = (const float*)d_in[1];
  const float* pos = (const float*)d_in[2];
  const float* Wq  = (const float*)d_in[3];
  const float* Wk  = (const float*)d_in[4];
  const float* Wv  = (const float*)d_in[5];
  const float* W1  = (const float*)d_in[6];
  const float* b1  = (const float*)d_in[7];
  const float* W2  = (const float*)d_in[8];
  const float* b2  = (const float*)d_in[9];
  const float* g1  = (const float*)d_in[10];
  const float* bb1 = (const float*)d_in[11];
  const float* g2  = (const float*)d_in[12];
  const float* bb2 = (const float*)d_in[13];
  const float* Wc  = (const float*)d_in[14];
  const float* bc  = (const float*)d_in[15];
  float* out = (float*)d_out;

  char* ws = (char*)d_ws;
  size_t off = 0;
  auto take = [&](size_t bytes) -> char* {
    char* p = ws + off;
    off = (off + bytes + 255) & ~(size_t)255;
    return p;
  };
  float* xf   = (float*)take((size_t)BQ * TT * DD * 4);   // fp32 activations
  bf16*  xb   = (bf16*) take((size_t)BQ * TT * DP * 2);   // padded bf16 mirror
  bf16*  qb   = (bf16*) take((size_t)BQ * HH * TT * KP * 2);
  bf16*  kb   = (bf16*) take((size_t)BQ * HH * TT * KP * 2);
  bf16*  vt   = (bf16*) take((size_t)BQ * HH * KP * SP * 2);  // V transposed
  bf16*  attn = (bf16*) take((size_t)BQ * HH * TT * SP * 2);
  float* scr  = (float*)take((size_t)BQ * TT * DP * 4);   // scores OR o OR ffn-out
  bf16*  hb   = (bf16*) take((size_t)BQ * TT * HKP * 2);  // ffn hidden bf16
  bf16*  wqp  = (bf16*) take((size_t)LL * HH * 6 * 17 * 512 * 2);
  bf16*  wkp  = (bf16*) take((size_t)LL * HH * 6 * 17 * 512 * 2);
  bf16*  wvp  = (bf16*) take((size_t)LL * HH * 6 * 17 * 512 * 2);
  bf16*  w1p  = (bf16*) take((size_t)LL * 34 * 17 * 512 * 2);
  bf16*  w2p  = (bf16*) take((size_t)LL * 34 * 17 * 512 * 2);

  // --- pack all weights (bf16, zero-padded, B-fragment lane tile layout) ---
  {
    int tq = LL * HH * 6 * 17 * 512;
    pack_w_kernel<<<(tq + 255) / 256, 256, 0, stream>>>(Wq, wqp, LL * HH, DD, KK, 17, 6);
    pack_w_kernel<<<(tq + 255) / 256, 256, 0, stream>>>(Wk, wkp, LL * HH, DD, KK, 17, 6);
    pack_w_kernel<<<(tq + 255) / 256, 256, 0, stream>>>(Wv, wvp, LL * HH, DD, KK, 17, 6);
    int tf = LL * 34 * 17 * 512;
    pack_w_kernel<<<(tf + 255) / 256, 256, 0, stream>>>(W1, w1p, LL, DD, HK, 17, 34);
    pack_w_kernel<<<(tf + 255) / 256, 256, 0, stream>>>(W2, w2p, LL, HK, DD, 17, 34);
  }

  // --- embedding ---
  {
    size_t tot = (size_t)BQ * TT * DP;
    embed_kernel<<<(int)((tot + 255) / 256), 256, 0, stream>>>(tok, emb, pos, xf, xb);
  }

  const size_t wqp_l = (size_t)HH * 6 * 17 * 512;  // per-layer packed QKV stride (elems)
  const size_t w1p_l = (size_t)34 * 17 * 512;      // per-layer packed FFN stride

  const int gqkv = (BQ * HH * 30 + 7) / 8;   // fused qkv tiles, 8 waves/block
  const int grow = (BQ * HH * 5 + 7) / 8;    // scores / attn@v row blocks
  const int st   = (BQ * HH * SP + 255) / 256;
  const int lr   = (BQ * TT + 7) / 8;
  const int gf   = (BQ * 5 * 17 + 7) / 8;    // ffn (2 n-tiles per wave)

  for (int i = 0; i < LL; ++i) {
    gemm_qkv_fused_kernel<<<gqkv, 256, 0, stream>>>(
        xb, wqp + i * wqp_l, wkp + i * wqp_l, wvp + i * wqp_l, qb, kb, vt);
    gemm_scores_kernel<<<grow, 256, 0, stream>>>(qb, kb, scr);
    softmax_kernel<<<st, 256, 0, stream>>>(scr, attn);
    gemm_av_kernel<<<grow, 256, 0, stream>>>(attn, vt, scr);          // scr now = o
    ln_kernel<<<lr, 256, 0, stream>>>(scr, xf, xb, g1 + i * DD, bb1 + i * DD);
    gemm_ffn1_kernel<<<gf, 256, 0, stream>>>(xb, w1p + i * w1p_l, b1 + i * HK, hb);
    gemm_ffn2_kernel<<<gf, 256, 0, stream>>>(hb, w2p + i * w1p_l, b2 + i * DD, scr);
    ln_kernel<<<lr, 256, 0, stream>>>(scr, xf, xb, g2 + i * DD, bb2 + i * DD);
  }

  final_kernel<<<BQ, 256, 0, stream>>>(xf, Wc, bc, out);
}
